// ALIGNNLayer_7275674599775
// MI455X (gfx1250) — matrix operand order repaired and checked by reading, not compiled, not run on hardware
//
#include <hip/hip_runtime.h>
#include <hip/hip_bf16.h>
#include <math.h>

#define NN 4096
#define NE 131072
#define NL 524288

typedef __attribute__((ext_vector_type(16))) __bf16 v16bf;
typedef __attribute__((ext_vector_type(8)))  float  v8f;

static constexpr int PSTR = 40;   // padded k-stride (80B: 16B-aligned, bank-conflict-free)

// ---------------------------------------------------------------------------
// WMMA + DPP helpers (CDNA5 wave32)
// ---------------------------------------------------------------------------
__device__ inline v8f wmma_bf16(v16bf a, v16bf b, v8f c) {
  return __builtin_amdgcn_wmma_f32_16x16x32_bf16(
      false, a, false, b, (short)0, c, false, false);
}

template <int CTRL>
__device__ inline float dpp_f(float x) {
  const int xi = __float_as_int(x);
  const int yi = __builtin_amdgcn_update_dpp(xi, xi, CTRL, 0xf, 0xf, true);
  return __int_as_float(yi);
}
// 16-lane butterfly all-reduce: xor1, xor2, xor7, xor15 — pure VALU DPP
__device__ inline float red16_max(float x) {
  x = fmaxf(x, dpp_f<0x0B1>(x));   // quad_perm [1,0,3,2]
  x = fmaxf(x, dpp_f<0x04E>(x));   // quad_perm [2,3,0,1]
  x = fmaxf(x, dpp_f<0x141>(x));   // row_half_mirror
  x = fmaxf(x, dpp_f<0x140>(x));   // row_mirror
  return x;
}
__device__ inline float red16_sum(float x) {
  x += dpp_f<0x0B1>(x);
  x += dpp_f<0x04E>(x);
  x += dpp_f<0x141>(x);
  x += dpp_f<0x140>(x);
  return x;
}
// 4-lane (quad) sum for LayerNorm partials
__device__ inline float red4_sum(float x) {
  x += dpp_f<0x0B1>(x);
  x += dpp_f<0x04E>(x);
  return x;
}

// A fragment: 16x32 bf16, row-major source, ld elements/row (ld*2 % 16 == 0).
// Per lane: two contiguous 16B chunks -> 2x ds_load_b128.
__device__ inline v16bf load_a_frag(const __bf16* src, int ld) {
  const int lane = threadIdx.x & 31;
  const int row  = lane & 15;
  const int kb   = (lane >> 4) << 4;
  const __bf16* p = src + row * ld + kb;
  v16bf a;
#pragma unroll
  for (int i = 0; i < 8; ++i) a[i] = p[i];
#pragma unroll
  for (int i = 0; i < 8; ++i) a[8 + i] = p[8 + i];
  return a;
}

// B fragment from a TRANSPOSED panel: element (k, col) at src[col*PSTR + k].
// Per lane: 16 contiguous bf16 -> 2x ds_load_b128, conflict-free.
__device__ inline v16bf load_b_panelT(const __bf16* src) {
  const int lane = threadIdx.x & 31;
  const int col  = lane & 15;
  const int kb   = (lane >> 4) << 4;
  const __bf16* p = src + col * PSTR + kb;
  v16bf b;
#pragma unroll
  for (int i = 0; i < 16; ++i) b[i] = p[i];
  return b;
}

// ---------------------------------------------------------------------------
// Fused gather -> Linear(K1,HID) -> SiLU -> Linear(HID,128) -> +resid -> LN
// Block: 128 threads (4 waves), 2 row-tiles x 2 col-groups, 32 rows/block.
// MODE 0: line  x=[ef[ls], ef[ld]]     MODE 1: edge  x=[ef, ef, aggr/cnt]
// MODE 2: node  x=[nf, aggr/cnt]
// ---------------------------------------------------------------------------
template <int K1, int HID, int MODE>
__global__ __launch_bounds__(128)
void fused_mlp_ln(const float* __restrict__ feat,
                  const int*   __restrict__ gatherS,
                  const int*   __restrict__ gatherD,
                  const float* __restrict__ aggr,
                  const float* __restrict__ cnt,
                  const float* __restrict__ w1, const float* __restrict__ b1,
                  const float* __restrict__ w2, const float* __restrict__ b2,
                  const float* __restrict__ resid,
                  const float* __restrict__ lng, const float* __restrict__ lnb,
                  float* __restrict__ outp)
{
  constexpr int XSTR = K1 + 8;     // padded strides (bytes % 16 == 0)
  constexpr int HSTR = HID + 8;
  constexpr int OSTR = 132;

  extern __shared__ char smem[];
  __bf16* sX = (__bf16*)smem;                                        // [32][XSTR]
  __bf16* sP = (__bf16*)(smem + (size_t)32 * XSTR * 2);              // [HID][PSTR] transposed
  __bf16* sH = (__bf16*)(smem + (size_t)32 * XSTR * 2
                              + (size_t)HID * PSTR * 2);             // [32][HSTR]
  float*  sO = (float*)smem;                                         // [32][OSTR] aliases sX

  const int tid  = threadIdx.x;
  const int wave = tid >> 5;
  const int lane = tid & 31;
  const int rt   = wave >> 1;
  const int cg   = wave & 1;
  const int base = blockIdx.x * 32;

  // prefetch the residual stream (lowers to global_prefetch_b8)
  if (tid < 32) __builtin_prefetch(resid + (long)(base + tid) * 128, 0, 1);

  // ---- gather + fp32->bf16 (float4 global loads) ----
  for (int e = tid; e < 32 * (K1 / 4); e += 128) {
    const int rl  = e / (K1 / 4);
    const int col = (e - rl * (K1 / 4)) * 4;
    const int r   = base + rl;
    const int seg = col >> 7;
    const int c   = col & 127;
    float4 v;
    if constexpr (MODE == 0) {
      const int src = (seg == 0) ? gatherS[r] : gatherD[r];
      v = *(const float4*)(feat + (long)src * 128 + c);
    } else if constexpr (MODE == 1) {
      if (seg < 2) v = *(const float4*)(feat + (long)r * 128 + c);
      else {
        v = *(const float4*)(aggr + (long)r * 128 + c);
        const float s = 1.0f / fmaxf(cnt[r], 1.0f);
        v.x *= s; v.y *= s; v.z *= s; v.w *= s;
      }
    } else {
      if (seg == 0) v = *(const float4*)(feat + (long)r * 128 + c);
      else {
        v = *(const float4*)(aggr + (long)r * 128 + c);
        const float s = 1.0f / fmaxf(cnt[r], 1.0f);
        v.x *= s; v.y *= s; v.z *= s; v.w *= s;
      }
    }
    __bf16* d = sX + rl * XSTR + col;
    d[0] = (__bf16)v.x; d[1] = (__bf16)v.y; d[2] = (__bf16)v.z; d[3] = (__bf16)v.w;
  }

  // ---- GEMM1: [16 x K1] @ [K1 x HID], transposed k-panels of 32 in LDS ----
  constexpr int NT1 = HID / 32;
  v8f acc1[NT1];
#pragma unroll
  for (int t = 0; t < NT1; ++t) {
    const float bv = b1[(cg * NT1 + t) * 16 + (lane & 15)];
#pragma unroll
    for (int i = 0; i < 8; ++i) acc1[t][i] = bv;
  }

#pragma unroll 1
  for (int kc = 0; kc < K1 / 32; ++kc) {
    __syncthreads();
    for (int e = tid; e < (32 * HID) / 4; e += 128) {
      const int kk = e / (HID / 4);
      const int n  = (e - kk * (HID / 4)) * 4;
      const float4 wv = *(const float4*)(w1 + (long)(kc * 32 + kk) * HID + n);
      sP[(n + 0) * PSTR + kk] = (__bf16)wv.x;
      sP[(n + 1) * PSTR + kk] = (__bf16)wv.y;
      sP[(n + 2) * PSTR + kk] = (__bf16)wv.z;
      sP[(n + 3) * PSTR + kk] = (__bf16)wv.w;
    }
    __syncthreads();
    const v16bf a = load_a_frag(sX + rt * 16 * XSTR + kc * 32, XSTR);
#pragma unroll
    for (int t = 0; t < NT1; ++t) {
      const v16bf b = load_b_panelT(sP + (cg * NT1 + t) * 16 * PSTR);
      acc1[t] = wmma_bf16(a, b, acc1[t]);
    }
  }

  // ---- SiLU -> sH (bf16) ----
  {
    const int rb = (lane >> 4) << 3;
    const int c0 = lane & 15;
#pragma unroll
    for (int t = 0; t < NT1; ++t) {
      const int col = (cg * NT1 + t) * 16 + c0;
#pragma unroll
      for (int i = 0; i < 8; ++i) {
        const float h = acc1[t][i];
        sH[(rt * 16 + rb + i) * HSTR + col] = (__bf16)(h / (1.0f + __expf(-h)));
      }
    }
  }
  __syncthreads();

  // ---- GEMM2: [16 x HID] @ [HID x 128] ----
  constexpr int NT2 = 4;
  v8f acc2[NT2];
#pragma unroll
  for (int t = 0; t < NT2; ++t) {
    const float bv = b2[(cg * NT2 + t) * 16 + (lane & 15)];
#pragma unroll
    for (int i = 0; i < 8; ++i) acc2[t][i] = bv;
  }

#pragma unroll 1
  for (int kc = 0; kc < HID / 32; ++kc) {
    __syncthreads();
    for (int e = tid; e < (32 * 128) / 4; e += 128) {
      const int kk = e >> 5;
      const int n  = (e & 31) * 4;
      const float4 wv = *(const float4*)(w2 + (long)(kc * 32 + kk) * 128 + n);
      sP[(n + 0) * PSTR + kk] = (__bf16)wv.x;
      sP[(n + 1) * PSTR + kk] = (__bf16)wv.y;
      sP[(n + 2) * PSTR + kk] = (__bf16)wv.z;
      sP[(n + 3) * PSTR + kk] = (__bf16)wv.w;
    }
    __syncthreads();
    const v16bf a = load_a_frag(sH + rt * 16 * HSTR + kc * 32, HSTR);
#pragma unroll
    for (int t = 0; t < NT2; ++t) {
      const v16bf b = load_b_panelT(sP + (cg * NT2 + t) * 16 * PSTR);
      acc2[t] = wmma_bf16(a, b, acc2[t]);
    }
  }
  __syncthreads();   // all reads of sX finished; safe to alias sO

  // ---- spill tile to LDS (fp32) + residual ----
  {
    const int rb = (lane >> 4) << 3;
    const int c0 = lane & 15;
#pragma unroll
    for (int t = 0; t < NT2; ++t) {
      const int col = (cg * NT2 + t) * 16 + c0;
#pragma unroll
      for (int i = 0; i < 8; ++i)
        sO[(rt * 16 + rb + i) * OSTR + col] = acc2[t][i];
    }
  }
  __syncthreads();
  for (int e = tid; e < 32 * 32; e += 128) {
    const int rl  = e >> 5;
    const int col = (e & 31) * 4;
    const float4 rv = *(const float4*)(resid + (long)(base + rl) * 128 + col);
    float* o = sO + rl * OSTR + col;
    o[0] += rv.x; o[1] += rv.y; o[2] += rv.z; o[3] += rv.w;
  }
  __syncthreads();

  // ---- LayerNorm: 4 lanes per row, quad DPP reduction ----
  {
    const int row = tid >> 2;
    const int sub = tid & 3;
    const float* ro = sO + row * OSTR;
    float s = 0.f;
#pragma unroll
    for (int c = 0; c < 32; ++c) s += ro[sub * 32 + c];
    s = red4_sum(s);
    const float mean = s * (1.0f / 128.0f);
    float v = 0.f;
#pragma unroll
    for (int c = 0; c < 32; ++c) { const float d = ro[sub * 32 + c] - mean; v += d * d; }
    v = red4_sum(v);
    const float rstd = rsqrtf(v * (1.0f / 128.0f) + 1e-5f);
    const long gr = base + row;
#pragma unroll
    for (int c = 0; c < 32; ++c) {
      const int cc = sub * 32 + c;
      outp[gr * 128 + cc] = (ro[cc] - mean) * rstd * lng[cc] + lnb[cc];
    }
  }
}

// ---------------------------------------------------------------------------
// Scatter-add (segment sum + count)
// ---------------------------------------------------------------------------
__global__ void scatter_add_kernel(const float* __restrict__ vals,
                                   const int* __restrict__ idx, int nrows,
                                   float* __restrict__ accum,
                                   float* __restrict__ cnt)
{
  const long t = (long)blockIdx.x * blockDim.x + threadIdx.x;
  if (t >= (long)nrows * 128) return;
  const int r = (int)(t >> 7);
  const int c = (int)(t & 127);
  const int d = idx[r];
  atomicAdd(&accum[(long)d * 128 + c], vals[t]);
  if (c == 0) atomicAdd(&cnt[d], 1.0f);
}

// ---------------------------------------------------------------------------
// GEMM + bias: A[rows x 128] @ W[128 x N] + b
// ---------------------------------------------------------------------------
template <int N>
__global__ __launch_bounds__(128)
void gemm_bias_kernel(const float* __restrict__ A, const float* __restrict__ W,
                      const float* __restrict__ bias, float* __restrict__ out)
{
  constexpr int NT   = N / 32;
  constexpr int ASTR = 136;
  __shared__ __bf16 sA[32 * ASTR];
  __shared__ __bf16 sP[N * PSTR];       // transposed panel

  const int tid  = threadIdx.x;
  const int wave = tid >> 5;
  const int lane = tid & 31;
  const int rt   = wave >> 1;
  const int cg   = wave & 1;
  const long base = (long)blockIdx.x * 32;

  for (int e = tid; e < 32 * 32; e += 128) {
    const int row = e >> 5;
    const int col = (e & 31) * 4;
    const float4 av = *(const float4*)(A + (base + row) * 128 + col);
    __bf16* d = sA + row * ASTR + col;
    d[0] = (__bf16)av.x; d[1] = (__bf16)av.y; d[2] = (__bf16)av.z; d[3] = (__bf16)av.w;
  }

  v8f acc[NT];
#pragma unroll
  for (int t = 0; t < NT; ++t) {
    const float bv = bias[(cg * NT + t) * 16 + (lane & 15)];
#pragma unroll
    for (int i = 0; i < 8; ++i) acc[t][i] = bv;
  }

#pragma unroll 1
  for (int kc = 0; kc < 4; ++kc) {
    __syncthreads();
    for (int e = tid; e < (32 * N) / 4; e += 128) {
      const int kk = e / (N / 4);
      const int n  = (e - kk * (N / 4)) * 4;
      const float4 wv = *(const float4*)(W + (long)(kc * 32 + kk) * N + n);
      sP[(n + 0) * PSTR + kk] = (__bf16)wv.x;
      sP[(n + 1) * PSTR + kk] = (__bf16)wv.y;
      sP[(n + 2) * PSTR + kk] = (__bf16)wv.z;
      sP[(n + 3) * PSTR + kk] = (__bf16)wv.w;
    }
    __syncthreads();
    const v16bf a = load_a_frag(sA + rt * 16 * ASTR + kc * 32, ASTR);
#pragma unroll
    for (int t = 0; t < NT; ++t) {
      const v16bf b = load_b_panelT(sP + (cg * NT + t) * 16 * PSTR);
      acc[t] = wmma_bf16(a, b, acc[t]);
    }
  }

  const int rb = (lane >> 4) << 3;
  const int c0 = lane & 15;
#pragma unroll
  for (int t = 0; t < NT; ++t) {
    const int col = (cg * NT + t) * 16 + c0;
#pragma unroll
    for (int i = 0; i < 8; ++i)
      out[(base + rt * 16 + rb + i) * N + col] = acc[t][i];
  }
}

// ---------------------------------------------------------------------------
// Flash attention: 4 heads, dh=32. qkv[r] = [q(128) k(128) v(128)].
// Block: 4 waves = 4 query tiles of one head; K/V (32 rows) staged in LDS
// (V transposed); online softmax with DPP reductions.
// ---------------------------------------------------------------------------
__global__ __launch_bounds__(128)
void attn_kernel(const float* __restrict__ qkv, float* __restrict__ ctx, int nq)
{
  constexpr int QSTR = 40;
  __shared__ __bf16 sQ[4][16 * QSTR];
  __shared__ __bf16 sK[32 * QSTR];      // [kv][dh]
  __shared__ __bf16 sVT[32 * QSTR];     // [dh][kv] transposed
  __shared__ __bf16 sPb[4][16 * QSTR];

  const int tid  = threadIdx.x;
  const int wave = tid >> 5;
  const int lane = tid & 31;
  const int h    = blockIdx.y;
  const int qbase = (blockIdx.x * 4 + wave) * 16;
  const float scale = 0.17677669529663689f;   // 1/sqrt(32)

  for (int e = lane; e < 512; e += 32) {
    const int row = e >> 5, c = e & 31;
    sQ[wave][row * QSTR + c] =
        (__bf16)(qkv[(long)(qbase + row) * 384 + h * 32 + c] * scale);
  }

  float m[8], l[8];
  v8f o0 = {}, o1 = {};
#pragma unroll
  for (int i = 0; i < 8; ++i) { m[i] = -1e30f; l[i] = 0.f; }

  const int rb = (lane >> 4) << 3;
  const int c0 = lane & 15;

  for (int kb = 0; kb < nq; kb += 32) {
    __syncthreads();
    for (int e = tid; e < 1024; e += 128) {
      const int kv = e >> 5, c = e & 31;
      const long ro = (long)(kb + kv) * 384 + h * 32;
      sK[kv * QSTR + c]  = (__bf16)qkv[ro + 128 + c];
      sVT[c * QSTR + kv] = (__bf16)qkv[ro + 256 + c];
    }
    __syncthreads();

    const v16bf a = load_a_frag(sQ[wave], QSTR);
    v8f s0 = {}, s1 = {};
    s0 = wmma_bf16(a, load_b_panelT(sK), s0);                  // (k=dh, col=kv 0..15)
    s1 = wmma_bf16(a, load_b_panelT(sK + 16 * QSTR), s1);      // kv 16..31

#pragma unroll
    for (int i = 0; i < 8; ++i) {
      const float mx = red16_max(fmaxf(s0[i], s1[i]));
      const float nm = fmaxf(m[i], mx);
      const float alpha = __expf(m[i] - nm);
      const float p0 = __expf(s0[i] - nm);
      const float p1 = __expf(s1[i] - nm);
      const float rs = red16_sum(p0 + p1);
      l[i] = l[i] * alpha + rs;
      o0[i] *= alpha; o1[i] *= alpha;
      m[i] = nm;
      sPb[wave][(rb + i) * QSTR + c0]      = (__bf16)p0;
      sPb[wave][(rb + i) * QSTR + 16 + c0] = (__bf16)p1;
    }

    const v16bf ap = load_a_frag(sPb[wave], QSTR);
    o0 = wmma_bf16(ap, load_b_panelT(sVT), o0);                // (k=kv, col=dh 0..15)
    o1 = wmma_bf16(ap, load_b_panelT(sVT + 16 * QSTR), o1);    // dh 16..31
  }

#pragma unroll
  for (int i = 0; i < 8; ++i) {
    const float inv = 1.0f / l[i];
    const long r = qbase + rb + i;
    ctx[r * 128 + h * 32 + c0]      = o0[i] * inv;
    ctx[r * 128 + h * 32 + 16 + c0] = o1[i] * inv;
  }
}

// ---------------------------------------------------------------------------
// Launcher
// ---------------------------------------------------------------------------
extern "C" void kernel_launch(void* const* d_in, const int* in_sizes, int n_in,
                              void* d_out, int out_size, void* d_ws, size_t ws_size,
                              hipStream_t stream) {
  (void)in_sizes; (void)n_in; (void)out_size; (void)ws_size;

  const float* nf   = (const float*)d_in[0];
  const float* ef   = (const float*)d_in[1];
  const float* lf   = (const float*)d_in[2];
  const int*   eidx = (const int*)d_in[3];
  const int*   lidx = (const int*)d_in[4];
  const float* le_w1 = (const float*)d_in[5];
  const float* le_b1 = (const float*)d_in[6];
  const float* le_w2 = (const float*)d_in[7];
  const float* le_b2 = (const float*)d_in[8];
  const float* e_w1  = (const float*)d_in[9];
  const float* e_b1  = (const float*)d_in[10];
  const float* e_w2  = (const float*)d_in[11];
  const float* e_b2  = (const float*)d_in[12];
  const float* n_w1  = (const float*)d_in[13];
  const float* n_b1  = (const float*)d_in[14];
  const float* n_w2  = (const float*)d_in[15];
  const float* n_b2  = (const float*)d_in[16];
  const float* wqkv  = (const float*)d_in[17];
  const float* bqkv  = (const float*)d_in[18];
  const float* wo    = (const float*)d_in[19];
  const float* bo    = (const float*)d_in[20];
  const float* ln_ng = (const float*)d_in[21];
  const float* ln_nb = (const float*)d_in[22];
  const float* ln_eg = (const float*)d_in[23];
  const float* ln_eb = (const float*)d_in[24];
  const float* ln_lg = (const float*)d_in[25];
  const float* ln_lb = (const float*)d_in[26];

  float* out_node = (float*)d_out;
  float* out_edge = out_node + (long)NN * 128;
  float* out_line = out_edge + (long)NE * 128;

  float* aggrE = (float*)d_ws;                 // NE*128
  float* cntE  = aggrE + (long)NE * 128;       // NE
  float* aggrN = cntE + NE;                    // NN*128
  float* cntN  = aggrN + (long)NN * 128;       // NN
  float* nupd  = cntN + NN;                    // NN*128
  float* qkvb  = nupd + (long)NN * 128;        // NN*384
  float* ctxb  = qkvb + (long)NN * 384;        // NN*128

  hipMemsetAsync(aggrE, 0,
                 ((size_t)NE * 128 + NE + (size_t)NN * 128 + NN) * sizeof(float),
                 stream);

  // 1) line-edge MLP + LN
  {
    const size_t smem = (size_t)32 * (256 + 8) * 2 + (size_t)128 * PSTR * 2
                      + (size_t)32 * (128 + 8) * 2;
    fused_mlp_ln<256, 128, 0><<<NL / 32, 128, smem, stream>>>(
        ef, lidx, lidx + NL, nullptr, nullptr,
        le_w1, le_b1, le_w2, le_b2, lf, ln_lg, ln_lb, out_line);
  }

  // 2) segment-mean line -> edges
  scatter_add_kernel<<<(int)(((long)NL * 128) / 256), 256, 0, stream>>>(
      out_line, lidx + NL, NL, aggrE, cntE);

  // 3) edge MLP + LN
  {
    const size_t smem = (size_t)32 * (384 + 8) * 2 + (size_t)256 * PSTR * 2
                      + (size_t)32 * (256 + 8) * 2;
    fused_mlp_ln<384, 256, 1><<<NE / 32, 128, smem, stream>>>(
        ef, nullptr, nullptr, aggrE, cntE,
        e_w1, e_b1, e_w2, e_b2, ef, ln_eg, ln_eb, out_edge);
  }

  // 4) segment-mean edges -> nodes
  scatter_add_kernel<<<(int)(((long)NE * 128) / 256), 256, 0, stream>>>(
      out_edge, eidx + NE, NE, aggrN, cntN);

  // 5) node MLP + LN -> nupd
  {
    const size_t smem = (size_t)32 * (256 + 8) * 2 + (size_t)256 * PSTR * 2
                      + (size_t)32 * (256 + 8) * 2;
    fused_mlp_ln<256, 256, 2><<<NN / 32, 128, smem, stream>>>(
        nf, nullptr, nullptr, aggrN, cntN,
        n_w1, n_b1, n_w2, n_b2, nf, ln_ng, ln_nb, nupd);
  }

  // 6) QKV projection
  gemm_bias_kernel<384><<<NN / 32, 128, 0, stream>>>(nupd, wqkv, bqkv, qkvb);

  // 7) flash attention
  attn_kernel<<<dim3(NN / 64, 4), 128, 0, stream>>>(qkvb, ctxb, NN);

  // 8) output projection -> node output
  gemm_bias_kernel<128><<<NN / 32, 128, 0, stream>>>(ctxb, wo, bo, out_node);
}
